// MoEGemmaMLP_39092792329155
// MI455X (gfx1250) — matrix operand order repaired and checked by reading, not compiled, not run on hardware
//
#include <hip/hip_runtime.h>
#include <stdint.h>

// ---------------- types ----------------
typedef __bf16 bf16_t;
typedef __attribute__((ext_vector_type(8)))  bf16_t v8bf;
typedef __attribute__((ext_vector_type(16))) bf16_t v16bf;
typedef __attribute__((ext_vector_type(8)))  float  v8f;

// ---------------- problem sizes ----------------
#define H_DIM 1024
#define I_DIM 4096
#define B_SZ  32
#define S_SZ  256
#define M_TOT (B_SZ * S_SZ)   // 8192 tokens
#define NEXP  8
#define NCOL  9               // 8 experts + shared column

// ---------------- GEMM tiling ----------------
#define BM 64
#define BN 128
#define BK 32
#define LDA 40    // padded LDS row stride in ushorts (80B)
#define LDB 40
#define NTHREADS 256

static __device__ __forceinline__ uint16_t f2bf(float f) {
  uint32_t x = __float_as_uint(f);
  uint32_t r = x + 0x7FFFu + ((x >> 16) & 1u);   // RNE
  return (uint16_t)(r >> 16);
}

// Branch-free gelu(tanh approx):
// tanh(u) = 1 - 2/(1 + e^{2u})  (saturates correctly at +/-inf)
static __device__ __forceinline__ float gelu_tanh(float x) {
  float u = 0.7978845608028654f * (x + 0.044715f * x * x * x);
  float e = __expf(2.0f * u);
  float t = 1.0f - 2.0f * __builtin_amdgcn_rcpf(e + 1.0f);
  return 0.5f * x * (1.0f + t);
}

// ---------------- CDNA5 async global->LDS (ASYNCcnt tracked) ----------------
// Each lane copies 16 bytes straight into LDS, no VGPR round-trip.
static __device__ __forceinline__ void async_b128(uint16_t* lds,
                                                  const uint16_t* g) {
  uint32_t l = (uint32_t)(uintptr_t)lds;   // low 32 bits = LDS byte offset
  asm volatile("global_load_async_to_lds_b128 %0, %1, off"
               :: "v"(l), "v"(g)
               : "memory");
}

static __device__ __forceinline__ void wait_async0() {
#if defined(__has_builtin)
#if __has_builtin(__builtin_amdgcn_s_wait_asynccnt)
  __builtin_amdgcn_s_wait_asynccnt(0);
#else
  asm volatile("s_wait_asynccnt 0x0" ::: "memory");
#endif
#else
  asm volatile("s_wait_asynccnt 0x0" ::: "memory");
#endif
}

// ---------------- small kernels ----------------
__global__ void router_kernel(const float* __restrict__ logits,
                              float* __restrict__ dw) {
  int b = threadIdx.x;
  if (b >= B_SZ) return;
  float p[NEXP];
  float mx = -1e30f;
  #pragma unroll
  for (int e = 0; e < NEXP; ++e) { p[e] = logits[b * NEXP + e]; mx = fmaxf(mx, p[e]); }
  float s = 0.f;
  #pragma unroll
  for (int e = 0; e < NEXP; ++e) { p[e] = expf(p[e] - mx); s += p[e]; }
  #pragma unroll
  for (int e = 0; e < NEXP; ++e) p[e] /= s;
  int a0 = 0;
  #pragma unroll
  for (int e = 1; e < NEXP; ++e) if (p[e] > p[a0]) a0 = e;
  int a1 = (a0 == 0) ? 1 : 0;
  #pragma unroll
  for (int e = 0; e < NEXP; ++e) if (e != a0 && p[e] > p[a1]) a1 = e;
  float t = p[a0] + p[a1];
  #pragma unroll
  for (int e = 0; e < NEXP; ++e) dw[b * NCOL + e] = 0.f;
  dw[b * NCOL + a0] = p[a0] / t;
  dw[b * NCOL + a1] = p[a1] / t;
  dw[b * NCOL + NEXP] = 1.0f;   // shared path weight
}

__global__ void cast_f32_bf16(const float* __restrict__ in,
                              uint16_t* __restrict__ out, int n4) {
  int i = blockIdx.x * blockDim.x + threadIdx.x;
  if (i >= n4) return;
  float4 f = ((const float4*)in)[i];
  ushort4 o;
  o.x = f2bf(f.x); o.y = f2bf(f.y); o.z = f2bf(f.z); o.w = f2bf(f.w);
  ((ushort4*)out)[i] = o;
}

// in: float [Z][R][C]  ->  out: bf16 [Z][C][R]   (R, C multiples of 32)
__global__ void transpose_cast(const float* __restrict__ in,
                               uint16_t* __restrict__ out, int R, int C) {
  __shared__ float tile[32][33];
  size_t zoff = (size_t)blockIdx.z * (size_t)R * (size_t)C;
  const float* src = in + zoff;
  uint16_t* dst = out + zoff;
  int c0 = blockIdx.x * 32, r0 = blockIdx.y * 32;
  for (int i = threadIdx.y; i < 32; i += 8) {
    tile[i][threadIdx.x] = src[(size_t)(r0 + i) * C + (c0 + threadIdx.x)];
  }
  __syncthreads();
  for (int i = threadIdx.y; i < 32; i += 8) {
    dst[(size_t)(c0 + i) * R + (r0 + threadIdx.x)] = f2bf(tile[threadIdx.x][i]);
  }
}

// ---------------- fragment helpers ----------------
// A tile: row-major [BM][LDA]; 16x32 bf16 fragment (lanes 0-15
// rows, K 0..15; lanes 16-31 same rows, K 16..31)
static __device__ __forceinline__ v16bf load_frag_a(const uint16_t* smem,
                                                    int rowbase, int lane) {
  int r  = lane & 15;
  int kh = lane >> 4;
  const bf16_t* p = (const bf16_t*)(smem + (rowbase + r) * LDA);
  v8bf lo = *(const v8bf*)(p + kh * 16);
  v8bf hi = *(const v8bf*)(p + kh * 16 + 8);
  v16bf a;
  #pragma unroll
  for (int j = 0; j < 8; ++j) { a[j] = lo[j]; a[8 + j] = hi[j]; }
  return a;
}

// B tile staged K-contiguous: [BN][LDB] (row = output column n, 32 K values)
// 32x16 fragment: lane = column, kh selects K 0..15 / 16..31
static __device__ __forceinline__ v16bf load_frag_b(const uint16_t* smem,
                                                    int colbase, int lane) {
  int c  = lane & 15;
  int kh = lane >> 4;
  const bf16_t* p = (const bf16_t*)(smem + (colbase + c) * LDB + kh * 16);
  v8bf lo = *(const v8bf*)(p);
  v8bf hi = *(const v8bf*)(p + 8);
  v16bf b;
  #pragma unroll
  for (int j = 0; j < 8; ++j) { b[j] = lo[j]; b[8 + j] = hi[j]; }
  return b;
}

// stage 64x32 bf16 A tile (one async b128 per thread)
static __device__ __forceinline__ void stage_a(uint16_t* As,
                                               const uint16_t* g, int ld,
                                               int m0, int k0, int tid) {
  int row = tid >> 2;
  int col = (tid & 3) * 8;
  async_b128(As + row * LDA + col,
             g + (size_t)(m0 + row) * ld + k0 + col);
}

// stage 128x32 bf16 B tile (two async b128 per thread)
static __device__ __forceinline__ void stage_b(uint16_t* Bs,
                                               const uint16_t* g, int ld,
                                               int n0, int k0, int tid) {
  int row = tid >> 2;
  int col = (tid & 3) * 8;
  #pragma unroll
  for (int h = 0; h < 2; ++h) {
    int r = row + h * 64;
    async_b128(Bs + r * LDB + col,
               g + (size_t)(n0 + r) * ld + k0 + col);
  }
}

// ---------------- stage A: H = gelu(X@G) * (X@U), bf16 out ----------------
__global__ __launch_bounds__(NTHREADS)
void moe_gateup_kernel(const uint16_t* __restrict__ Xbf,   // [M_TOT][H]
                       const uint16_t* __restrict__ GT,    // [I][H]
                       const uint16_t* __restrict__ UT,    // [I][H]
                       uint16_t* __restrict__ Hbuf,        // [M_TOT][I]
                       const float* __restrict__ dw, int ecol) {
  int m0 = blockIdx.y * BM;
  int n0 = blockIdx.x * BN;
  int batch = m0 / S_SZ;
  if (dw[batch * NCOL + ecol] == 0.0f) return;   // not routed: skip

  __shared__ __align__(16) uint16_t As[2][BM * LDA];
  __shared__ __align__(16) uint16_t BsG[2][BN * LDB];
  __shared__ __align__(16) uint16_t BsU[2][BN * LDB];

  int tid = threadIdx.x, lane = tid & 31, wave = tid >> 5;
  int nw = wave * 16;

  v8f zero = {0.f,0.f,0.f,0.f,0.f,0.f,0.f,0.f};
  v8f accG[4], accU[4];
  #pragma unroll
  for (int mt = 0; mt < 4; ++mt) { accG[mt] = zero; accU[mt] = zero; }

  // prologue: stage first K-tile
  stage_a(As[0], Xbf, H_DIM, m0, 0, tid);
  stage_b(BsG[0], GT, H_DIM, n0, 0, tid);
  stage_b(BsU[0], UT, H_DIM, n0, 0, tid);
  wait_async0();
  __syncthreads();

  const int KT = H_DIM / BK;
  for (int kt = 0; kt < KT; ++kt) {
    int cur = kt & 1, nxt = cur ^ 1;
    if (kt + 1 < KT) {                     // async-prefetch next tile
      int k0n = (kt + 1) * BK;
      stage_a(As[nxt], Xbf, H_DIM, m0, k0n, tid);
      stage_b(BsG[nxt], GT, H_DIM, n0, k0n, tid);
      stage_b(BsU[nxt], UT, H_DIM, n0, k0n, tid);
    }
    v16bf bG = load_frag_b(BsG[cur], nw, lane);
    v16bf bU = load_frag_b(BsU[cur], nw, lane);
    #pragma unroll
    for (int mt = 0; mt < 4; ++mt) {
      v16bf a = load_frag_a(As[cur], mt * 16, lane);
      accG[mt] = __builtin_amdgcn_wmma_f32_16x16x32_bf16(
          false, a, false, bG, (short)0, accG[mt], false, false);
      accU[mt] = __builtin_amdgcn_wmma_f32_16x16x32_bf16(
          false, a, false, bU, (short)0, accU[mt], false, false);
    }
    wait_async0();
    __syncthreads();
  }

  // C layout: VGPR v, lanes 0-15 -> M=v, N=lane; lanes 16-31 -> M=8+v
  int cN = lane & 15, rHalf = (lane >> 4) * 8;
  #pragma unroll
  for (int mt = 0; mt < 4; ++mt) {
    #pragma unroll
    for (int v = 0; v < 8; ++v) {
      int m = m0 + mt * 16 + rHalf + v;
      int n = n0 + nw + cN;
      float hval = gelu_tanh(accG[mt][v]) * accU[mt][v];
      Hbuf[(size_t)m * I_DIM + n] = f2bf(hval);
    }
  }
}

// ---------------- stage B: Out (+)= w * (H @ D) ----------------
__global__ __launch_bounds__(NTHREADS)
void moe_down_kernel(const uint16_t* __restrict__ Hbuf,  // [M_TOT][I]
                     const uint16_t* __restrict__ DT,    // [H][I]
                     float* __restrict__ Out,            // [M_TOT][H]
                     const float* __restrict__ dw, int ecol, int accumulate) {
  int m0 = blockIdx.y * BM;
  int n0 = blockIdx.x * BN;
  int batch = m0 / S_SZ;
  float w = dw[batch * NCOL + ecol];
  if (w == 0.0f) return;

  __shared__ __align__(16) uint16_t As[2][BM * LDA];
  __shared__ __align__(16) uint16_t Bs[2][BN * LDB];

  int tid = threadIdx.x, lane = tid & 31, wave = tid >> 5;
  int nw = wave * 16;

  v8f zero = {0.f,0.f,0.f,0.f,0.f,0.f,0.f,0.f};
  v8f acc[4];
  #pragma unroll
  for (int mt = 0; mt < 4; ++mt) acc[mt] = zero;

  stage_a(As[0], Hbuf, I_DIM, m0, 0, tid);
  stage_b(Bs[0], DT, I_DIM, n0, 0, tid);
  wait_async0();
  __syncthreads();

  const int KT = I_DIM / BK;
  for (int kt = 0; kt < KT; ++kt) {
    int cur = kt & 1, nxt = cur ^ 1;
    if (kt + 1 < KT) {
      int k0n = (kt + 1) * BK;
      stage_a(As[nxt], Hbuf, I_DIM, m0, k0n, tid);
      stage_b(Bs[nxt], DT, I_DIM, n0, k0n, tid);
    }
    v16bf b = load_frag_b(Bs[cur], nw, lane);
    #pragma unroll
    for (int mt = 0; mt < 4; ++mt) {
      v16bf a = load_frag_a(As[cur], mt * 16, lane);
      acc[mt] = __builtin_amdgcn_wmma_f32_16x16x32_bf16(
          false, a, false, b, (short)0, acc[mt], false, false);
    }
    wait_async0();
    __syncthreads();
  }

  int cN = lane & 15, rHalf = (lane >> 4) * 8;
  #pragma unroll
  for (int mt = 0; mt < 4; ++mt) {
    #pragma unroll
    for (int v = 0; v < 8; ++v) {
      size_t idx = (size_t)(m0 + mt * 16 + rHalf + v) * H_DIM + (n0 + nw + cN);
      float val = w * acc[mt][v];
      if (accumulate) Out[idx] += val;
      else            Out[idx]  = val;
    }
  }
}

// ---------------- host-side launch ----------------
extern "C" void kernel_launch(void* const* d_in, const int* in_sizes, int n_in,
                              void* d_out, int out_size, void* d_ws, size_t ws_size,
                              hipStream_t stream) {
  const float* x            = (const float*)d_in[0];  // [32,256,1024]
  const float* logits       = (const float*)d_in[1];  // [32,8]
  const float* skill_gate   = (const float*)d_in[2];  // [8,1024,4096]
  const float* skill_up     = (const float*)d_in[3];  // [8,1024,4096]
  const float* skill_down   = (const float*)d_in[4];  // [8,4096,1024]
  const float* shared_gate  = (const float*)d_in[5];  // [1024,4096]
  const float* shared_up    = (const float*)d_in[6];  // [1024,4096]
  const float* shared_down  = (const float*)d_in[7];  // [4096,1024]
  float* out = (float*)d_out;

  // ---- workspace layout (bytes, 256-aligned chunks) ----
  char* ws = (char*)d_ws;
  size_t off = 0;
  auto alloc = [&](size_t bytes) {
    char* p = ws + off;
    off += (bytes + 255) & ~(size_t)255;
    return p;
  };
  const size_t HI  = (size_t)H_DIM * I_DIM;            // 4,194,304
  float*    dw    = (float*)   alloc(B_SZ * NCOL * sizeof(float));
  uint16_t* Xbf   = (uint16_t*)alloc((size_t)M_TOT * H_DIM * 2);
  uint16_t* GTs   = (uint16_t*)alloc(NEXP * HI * 2);   // [e][I][H]
  uint16_t* UTs   = (uint16_t*)alloc(NEXP * HI * 2);   // [e][I][H]
  uint16_t* DTs   = (uint16_t*)alloc(NEXP * HI * 2);   // [e][H][I]
  uint16_t* GTsh  = (uint16_t*)alloc(HI * 2);
  uint16_t* UTsh  = (uint16_t*)alloc(HI * 2);
  uint16_t* DTsh  = (uint16_t*)alloc(HI * 2);
  uint16_t* Hbuf  = (uint16_t*)alloc((size_t)M_TOT * I_DIM * 2);
  (void)ws_size; (void)in_sizes; (void)n_in; (void)out_size;

  // 1) routing weights
  router_kernel<<<1, 32, 0, stream>>>(logits, dw);

  // 2) bf16 conversions
  {
    int n4 = (M_TOT * H_DIM) / 4;
    cast_f32_bf16<<<(n4 + 255) / 256, 256, 0, stream>>>(x, Xbf, n4);
  }
  dim3 tb(32, 8);
  // gate/up: [R=H][C=I] -> [I][H]
  transpose_cast<<<dim3(I_DIM / 32, H_DIM / 32, NEXP), tb, 0, stream>>>(skill_gate, GTs, H_DIM, I_DIM);
  transpose_cast<<<dim3(I_DIM / 32, H_DIM / 32, NEXP), tb, 0, stream>>>(skill_up,   UTs, H_DIM, I_DIM);
  // down: [R=I][C=H] -> [H][I]
  transpose_cast<<<dim3(H_DIM / 32, I_DIM / 32, NEXP), tb, 0, stream>>>(skill_down, DTs, I_DIM, H_DIM);
  transpose_cast<<<dim3(I_DIM / 32, H_DIM / 32, 1), tb, 0, stream>>>(shared_gate, GTsh, H_DIM, I_DIM);
  transpose_cast<<<dim3(I_DIM / 32, H_DIM / 32, 1), tb, 0, stream>>>(shared_up,   UTsh, H_DIM, I_DIM);
  transpose_cast<<<dim3(H_DIM / 32, I_DIM / 32, 1), tb, 0, stream>>>(shared_down, DTsh, I_DIM, H_DIM);

  // 3) MLPs: shared first (writes out), then routed experts (accumulate)
  dim3 gA(I_DIM / BN, M_TOT / BM);   // (32, 128)
  dim3 gB(H_DIM / BN, M_TOT / BM);   // (8, 128)

  moe_gateup_kernel<<<gA, NTHREADS, 0, stream>>>(Xbf, GTsh, UTsh, Hbuf, dw, NEXP);
  moe_down_kernel  <<<gB, NTHREADS, 0, stream>>>(Hbuf, DTsh, out, dw, NEXP, 0);

  for (int e = 0; e < NEXP; ++e) {
    moe_gateup_kernel<<<gA, NTHREADS, 0, stream>>>(Xbf, GTs + e * HI, UTs + e * HI,
                                                   Hbuf, dw, e);
    moe_down_kernel  <<<gB, NTHREADS, 0, stream>>>(Hbuf, DTs + e * HI, out, dw, e, 1);
  }
}
